// CausalAttention_73478300500353
// MI455X (gfx1250) — matrix operand order repaired and checked by reading, not compiled, not run on hardware
//
#include <hip/hip_runtime.h>

// CDNA5 / gfx1250 causal multi-head attention, f16 WMMA pipeline.
// B=4, L=2048, D=1024, H=16, dh=64.
// Round 3: attention key-chunks widened 32 -> 64 keys (halves softmax
// shuffle/VALU overhead per WMMA). Projections keep 32x64 tiles, all-f16
// operands via one-shot conversion pass.

#define L_SEQ 2048
#define D_MODEL 1024
#define N_HEADS 16
#define D_HEAD 64
#define BATCH 4
#define BH (BATCH * N_HEADS)      // 64
#define MROWS (BATCH * L_SEQ)     // 8192

typedef __attribute__((ext_vector_type(16))) _Float16 v16h;
typedef __attribute__((ext_vector_type(8)))  float    v8f;
typedef __attribute__((ext_vector_type(4)))  float    f4;
typedef __attribute__((ext_vector_type(8)))  _Float16 h8;

#define NEG_INF (-__builtin_inff())

__device__ __forceinline__ v8f wmma16(v16h a, v16h b, v8f c) {
  return __builtin_amdgcn_wmma_f32_16x16x32_f16(false, a, false, b, (short)0, c,
                                                false, false);
}

// ---- A-operand fragment (16x32 f16): lane row m = lane%16, half = lane/16.
// k elements: {8*half + 0..7} and {16 + 8*half + 0..7}.
// Pointer must be row_base + kbase + 8*half.
__device__ __forceinline__ v16h fragA_h16(const _Float16* p) {
  h8 g0 = *(const h8*)(p);
  h8 g1 = *(const h8*)(p + 16);
  v16h r;
#pragma unroll
  for (int i = 0; i < 8; ++i) { r[i] = g0[i]; r[8 + i] = g1[i]; }
  return r;
}

// ---- B-operand fragment (32x16 f16): lane col n = lane%16, half = lane/16.
// k elements: {8*half + 0..7} and {16 + 8*half + 0..7}. Pointer = row_base + kbase + 8*half.
__device__ __forceinline__ v16h fragB_h16(const _Float16* p) {
  h8 g0 = *(const h8*)(p);
  h8 g1 = *(const h8*)(p + 16);
  v16h r;
#pragma unroll
  for (int i = 0; i < 8; ++i) { r[i] = g0[i]; r[8 + i] = g1[i]; }
  return r;
}

// ============================================================================
// Kernel 0: one-shot f32 -> f16 conversion (8 elements / thread).
// ============================================================================
__global__ void k_cvt(const float* __restrict__ src, _Float16* __restrict__ dst) {
  const int i = (blockIdx.x * blockDim.x + threadIdx.x) * 8;
  f4 a = *(const f4*)(src + i);
  f4 b = *(const f4*)(src + i + 4);
  h8 o;
#pragma unroll
  for (int j = 0; j < 4; ++j) {
    o[j]     = (_Float16)a[j];
    o[4 + j] = (_Float16)b[j];
  }
  *(h8*)(dst + i) = o;
}

// ============================================================================
// Kernel 1: fused Q/K/V projections.  y = x @ W^T, all-f16 operands.
// Wave handles (proj, 32-row mtile, head): 32x64 output tile, K-loop of 32.
// Q,K stored [bh][L][64]; V stored transposed [bh][64][L].
// ============================================================================
__global__ void k_qkv(const _Float16* __restrict__ xh,
                      const _Float16* __restrict__ Wq,
                      const _Float16* __restrict__ Wk,
                      const _Float16* __restrict__ Wv,
                      _Float16* __restrict__ Q, _Float16* __restrict__ K,
                      _Float16* __restrict__ Vt) {
  const int tid = threadIdx.x;
  const int wave = tid >> 5, lane = tid & 31;
  const int half = lane >> 4, ln = lane & 15;
  const int gw = blockIdx.x * 8 + wave;          // 0 .. 3*256*16-1
  const int p   = gw / (256 * N_HEADS);          // 0=Q 1=K 2=V
  const int rem = gw % (256 * N_HEADS);
  const int mt  = rem >> 4;                      // 0..255 (32 rows each)
  const int h   = rem & 15;
  const _Float16* W = (p == 0) ? Wq : (p == 1) ? Wk : Wv;
  const _Float16* ar0 = xh + (size_t)(mt * 32 + ln) * D_MODEL;
  const _Float16* ar1 = xh + (size_t)(mt * 32 + 16 + ln) * D_MODEL;

  v8f acc[2][4] = {};
  for (int kc = 0; kc < D_MODEL; kc += 32) {
    v16h a0 = fragA_h16(ar0 + kc + half * 8);
    v16h a1 = fragA_h16(ar1 + kc + half * 8);
#pragma unroll
    for (int j = 0; j < 4; ++j) {
      v16h b = fragB_h16(W + (size_t)(h * 64 + j * 16 + ln) * D_MODEL + kc +
                         half * 8);
      acc[0][j] = wmma16(a0, b, acc[0][j]);
      acc[1][j] = wmma16(a1, b, acc[1][j]);
    }
  }

  const int b = mt >> 6;                          // batch (mt*32/2048)
  if (p < 2) {
    _Float16* dst = (p == 0) ? Q : K;
    const size_t base = (size_t)(b * N_HEADS + h) * L_SEQ * D_HEAD;
#pragma unroll
    for (int i = 0; i < 2; ++i) {
      const int m0 = (mt & 63) * 32 + i * 16 + 8 * half;
#pragma unroll
      for (int j = 0; j < 4; ++j)
#pragma unroll
        for (int r = 0; r < 8; ++r)
          dst[base + (size_t)(m0 + r) * D_HEAD + j * 16 + ln] =
              (_Float16)acc[i][j][r];
    }
  } else {
    const size_t base = (size_t)(b * N_HEADS + h) * D_HEAD * L_SEQ;
#pragma unroll
    for (int i = 0; i < 2; ++i) {
      const int m0 = (mt & 63) * 32 + i * 16 + 8 * half;
#pragma unroll
      for (int j = 0; j < 4; ++j) {
        h8 vv;
#pragma unroll
        for (int r = 0; r < 8; ++r) vv[r] = (_Float16)acc[i][j][r];
        *(h8*)(Vt + base + (size_t)(j * 16 + ln) * L_SEQ + m0) = vv;
      }
    }
  }
}

// ============================================================================
// Kernel 2: causal flash attention per (bh, 16-query tile), 64-key chunks.
// S = Q K^T (8 WMMAs/chunk), one online-softmax pass per 64 keys,
// P re-layout D-frag -> A-frag via LDS, O += P V^T (8 WMMAs/chunk).
// ============================================================================
__global__ void k_attn(const _Float16* __restrict__ Q,
                       const _Float16* __restrict__ K,
                       const _Float16* __restrict__ Vt,
                       _Float16* __restrict__ AO) {
  __shared__ _Float16 Pl[8][16 * 64];   // per-wave 16x64 P tile (16 KB)
  const int tid = threadIdx.x;
  const int wave = tid >> 5, lane = tid & 31;
  const int half = lane >> 4, ln = lane & 15;
  const int gw = blockIdx.x * 8 + wave;   // 0..8191
  const int bh = gw >> 7;                 // 0..63
  const int qt = gw & 127;                // query tile

  const _Float16* Qp = Q  + (size_t)bh * L_SEQ * D_HEAD;
  const _Float16* Kp = K  + (size_t)bh * L_SEQ * D_HEAD;
  const _Float16* Vp = Vt + (size_t)bh * D_HEAD * L_SEQ;

  // Q fragments (A-operand), reused across all key chunks.
  const _Float16* qrow = Qp + (size_t)(qt * 16 + ln) * D_HEAD;
  v16h qf0 = fragA_h16(qrow + half * 8);        // k = 0..31
  v16h qf1 = fragA_h16(qrow + 32 + half * 8);   // k = 32..63

  float mrow[8], lrow[8];
  v8f o[4] = {};
#pragma unroll
  for (int r = 0; r < 8; ++r) { mrow[r] = NEG_INF; lrow[r] = 0.f; }

  const int qend = qt * 16 + 15;
  const int nchunks = (qend >> 6) + 1;    // 64-key chunks, causal-truncated
  const float scale = 0.125f;             // 1/sqrt(64)

  for (int kc = 0; kc < nchunks; ++kc) {
    const int kbase = kc * 64;
    // ---- scores S (16 queries x 64 keys), four 16x16 N-tiles x two K-steps
    v8f s[4] = {};
#pragma unroll
    for (int nt = 0; nt < 4; ++nt) {
      const _Float16* krow = Kp + (size_t)(kbase + nt * 16 + ln) * D_HEAD;
      s[nt] = wmma16(qf0, fragB_h16(krow + half * 8), s[nt]);
      s[nt] = wmma16(qf1, fragB_h16(krow + 32 + half * 8), s[nt]);
    }
    // ---- scale + causal mask + one online-softmax pass for 64 keys
    const int qbase = qt * 16 + 8 * half;
#pragma unroll
    for (int r = 0; r < 8; ++r) {
      const int qq = qbase + r;
      float v[4];
#pragma unroll
      for (int nt = 0; nt < 4; ++nt) {
        const int key = kbase + nt * 16 + ln;
        v[nt] = (key <= qq) ? s[nt][r] * scale : NEG_INF;
      }
      float mx = fmaxf(fmaxf(v[0], v[1]), fmaxf(v[2], v[3]));
#pragma unroll
      for (int off = 8; off; off >>= 1) mx = fmaxf(mx, __shfl_xor(mx, off, 32));
      const float mn = fmaxf(mrow[r], mx);
      const float alpha = __expf(mrow[r] - mn);
      float rs = 0.f;
#pragma unroll
      for (int nt = 0; nt < 4; ++nt) {
        const float pv = __expf(v[nt] - mn);
        rs += pv;
        // P in D-frag layout -> LDS row-major [m][n], m = r+8*half
        Pl[wave][(r + 8 * half) * 64 + nt * 16 + ln] = (_Float16)pv;
      }
#pragma unroll
      for (int off = 8; off; off >>= 1) rs += __shfl_xor(rs, off, 32);
      lrow[r] = lrow[r] * alpha + rs;
      mrow[r] = mn;
#pragma unroll
      for (int j = 0; j < 4; ++j) o[j][r] *= alpha;
    }
    asm volatile("s_wait_dscnt 0" ::: "memory");
    // ---- O += P * V^T  (P as A-operand from LDS, Vt rows as B-operand)
    v16h pf0 = fragA_h16(&Pl[wave][ln * 64 + half * 8]);
    v16h pf1 = fragA_h16(&Pl[wave][ln * 64 + 32 + half * 8]);
#pragma unroll
    for (int j = 0; j < 4; ++j) {
      const _Float16* vrow = Vp + (size_t)(j * 16 + ln) * L_SEQ + kbase;
      o[j] = wmma16(pf0, fragB_h16(vrow + half * 8), o[j]);
      o[j] = wmma16(pf1, fragB_h16(vrow + 32 + half * 8), o[j]);
    }
  }

  // ---- normalize and store attn output as f16 [B*L][D] (heads concatenated)
  const int b = bh >> 4, h = bh & 15;
  const size_t obase =
      ((size_t)b * L_SEQ + qt * 16) * D_MODEL + h * D_HEAD;
#pragma unroll
  for (int j = 0; j < 4; ++j)
#pragma unroll
    for (int r = 0; r < 8; ++r) {
      float val = o[j][r] / lrow[r];
      AO[obase + (size_t)(r + 8 * half) * D_MODEL + j * 16 + ln] =
          (_Float16)val;
    }
}

// ============================================================================
// Kernel 3: output projection  out = AO @ Wo^T  (f16 x f16 -> f32),
// 32x64 tile per wave.
// ============================================================================
__global__ void k_out(const _Float16* __restrict__ AO,
                      const _Float16* __restrict__ Wo,
                      float* __restrict__ out) {
  const int tid = threadIdx.x;
  const int wave = tid >> 5, lane = tid & 31;
  const int half = lane >> 4, ln = lane & 15;
  const int gw = blockIdx.x * 8 + wave;   // 0..4095
  const int mt = gw >> 4;                 // 0..255 (32 rows each)
  const int nt = gw & 15;                 // 0..15 (64 cols each)
  const _Float16* ar0 = AO + (size_t)(mt * 32 + ln) * D_MODEL;
  const _Float16* ar1 = AO + (size_t)(mt * 32 + 16 + ln) * D_MODEL;

  v8f acc[2][4] = {};
  for (int kc = 0; kc < D_MODEL; kc += 32) {
    v16h a0 = fragA_h16(ar0 + kc + half * 8);
    v16h a1 = fragA_h16(ar1 + kc + half * 8);
#pragma unroll
    for (int j = 0; j < 4; ++j) {
      v16h b = fragB_h16(Wo + (size_t)(nt * 64 + j * 16 + ln) * D_MODEL + kc +
                         half * 8);
      acc[0][j] = wmma16(a0, b, acc[0][j]);
      acc[1][j] = wmma16(a1, b, acc[1][j]);
    }
  }
#pragma unroll
  for (int i = 0; i < 2; ++i)
#pragma unroll
    for (int j = 0; j < 4; ++j)
#pragma unroll
      for (int r = 0; r < 8; ++r)
        out[(size_t)(mt * 32 + i * 16 + r + 8 * half) * D_MODEL + nt * 64 +
            j * 16 + ln] = acc[i][j][r];
}

extern "C" void kernel_launch(void* const* d_in, const int* in_sizes, int n_in,
                              void* d_out, int out_size, void* d_ws,
                              size_t ws_size, hipStream_t stream) {
  const float* x  = (const float*)d_in[0];
  const float* Wq = (const float*)d_in[1];
  const float* Wk = (const float*)d_in[2];
  const float* Wv = (const float*)d_in[3];
  const float* Wo = (const float*)d_in[4];
  float* out = (float*)d_out;

  char* ws = (char*)d_ws;
  const size_t SZ  = (size_t)BH * L_SEQ * D_HEAD * sizeof(_Float16);  // 16 MB
  const size_t WSZ = (size_t)D_MODEL * D_MODEL * sizeof(_Float16);    //  2 MB
  _Float16* Qh  = (_Float16*)(ws);
  _Float16* Kh  = (_Float16*)(ws + SZ);
  _Float16* Vt  = (_Float16*)(ws + 2 * SZ);
  _Float16* AO  = (_Float16*)(ws + 3 * SZ);
  _Float16* xh  = (_Float16*)(ws + 4 * SZ);
  _Float16* Wqh = (_Float16*)(ws + 5 * SZ);
  _Float16* Wkh = (_Float16*)(ws + 5 * SZ + WSZ);
  _Float16* Wvh = (_Float16*)(ws + 5 * SZ + 2 * WSZ);
  _Float16* Woh = (_Float16*)(ws + 5 * SZ + 3 * WSZ);

  // one-shot f32 -> f16 conversions
  k_cvt<<<(MROWS * D_MODEL) / (8 * 256), 256, 0, stream>>>(x, xh);
  k_cvt<<<(D_MODEL * D_MODEL) / (8 * 256), 256, 0, stream>>>(Wq, Wqh);
  k_cvt<<<(D_MODEL * D_MODEL) / (8 * 256), 256, 0, stream>>>(Wk, Wkh);
  k_cvt<<<(D_MODEL * D_MODEL) / (8 * 256), 256, 0, stream>>>(Wv, Wvh);
  k_cvt<<<(D_MODEL * D_MODEL) / (8 * 256), 256, 0, stream>>>(Wo, Woh);

  // 3 * 256 * 16 = 12288 waves, 8 waves/block
  k_qkv<<<1536, 256, 0, stream>>>(xh, Wqh, Wkh, Wvh, Qh, Kh, Vt);
  // 64 * 128 = 8192 waves
  k_attn<<<1024, 256, 0, stream>>>(Qh, Kh, Vt, AO);
  // 256 * 16 = 4096 waves
  k_out<<<512, 256, 0, stream>>>(AO, Woh, out);
}
